// MultiHeadDotProductAttention_31275951849698
// MI455X (gfx1250) — matrix operand order repaired and checked by reading, compiled, not run on hardware
//
#include <hip/hip_runtime.h>
#include <cmath>

typedef __attribute__((ext_vector_type(16))) __bf16 v16bf;
typedef __attribute__((ext_vector_type(8)))  __bf16 v8bf;
typedef __attribute__((ext_vector_type(4)))  __bf16 v4bf;
typedef __attribute__((ext_vector_type(8)))  float  v8f;
typedef __attribute__((ext_vector_type(4)))  float  v4f;

// Problem constants (match reference)
constexpr int Bsz = 2, Tq = 2048, Ssz = 2048, Emb = 2048, NH = 16, DH = 128;
constexpr int Mrows  = Bsz * Tq;   // 4096
constexpr int KdimE  = Emb;        // 2048 reduction dim (all GEMMs)
constexpr int NcolsP = NH * DH;    // 2048 output cols  (all GEMMs)

static __device__ __forceinline__ __bf16 f32_to_bf16(float f) {
  unsigned u = __builtin_bit_cast(unsigned, f);
  u += 0x7fffu + ((u >> 16) & 1u);   // round to nearest even
  unsigned short h = (unsigned short)(u >> 16);
  return __builtin_bit_cast(__bf16, h);
}

static __device__ __forceinline__ v8f zero8() {
  v8f z;
#pragma unroll
  for (int i = 0; i < 8; ++i) z[i] = 0.0f;
  return z;
}

static __device__ __forceinline__ v16bf load_frag(const __bf16* p0, const __bf16* p1) {
  v8bf lo = *(const v8bf*)p0;
  v8bf hi = *(const v8bf*)p1;
  v16bf r;
#pragma unroll
  for (int i = 0; i < 8; ++i) { r[i] = lo[i]; r[i + 8] = hi[i]; }
  return r;
}

static __device__ __forceinline__ v8f wmma_bf16(v16bf a, v16bf b, v8f c) {
  return __builtin_amdgcn_wmma_f32_16x16x32_bf16(false, a, false, b, (short)0, c,
                                                 false, false);
}

// ---- CDNA5 async memory->LDS copy (ASYNCcnt-tracked), inline asm ------------
static __device__ __forceinline__ void async_b128(void* lds_ptr, const void* gptr) {
  unsigned l = (unsigned)(unsigned long long)lds_ptr;
  unsigned long long g = (unsigned long long)gptr;
  asm volatile("global_load_async_to_lds_b128 %0, %1, off"
               :: "v"(l), "v"(g) : "memory");
}
template <int N>
static __device__ __forceinline__ void wait_asynccnt() {
  asm volatile("s_wait_asynccnt %0" :: "i"(N) : "memory");
}

// ---------------------------------------------------------------- conversion
__global__ void __launch_bounds__(256) cvt_f32_bf16(const float* __restrict__ src,
                                                    __bf16* __restrict__ dst, int n4) {
  int i = blockIdx.x * 256 + threadIdx.x;
  if (i >= n4) return;
  v4f v = *(const v4f*)(src + (size_t)i * 4);
  v4bf o;
#pragma unroll
  for (int e = 0; e < 4; ++e) o[e] = f32_to_bf16(v[e]);
  *(v4bf*)(dst + (size_t)i * 4) = o;
}

// Transpose-convert a 2048x2048 f32 matrix into bf16: dst[c][r] = src[r][c].
__global__ void __launch_bounds__(256) cvt_transpose_bf16(const float* __restrict__ src,
                                                          __bf16* __restrict__ dst) {
  __shared__ __bf16 t[32][33];
  const int tx = threadIdx.x & 31, ty = threadIdx.x >> 5;   // 32x8
  const int r0 = blockIdx.y * 32, c0 = blockIdx.x * 32;
#pragma unroll
  for (int r = ty; r < 32; r += 8)
    t[r][tx] = f32_to_bf16(src[(size_t)(r0 + r) * 2048 + c0 + tx]);
  __syncthreads();
#pragma unroll
  for (int r = ty; r < 32; r += 8)
    dst[(size_t)(c0 + r) * 2048 + r0 + tx] = t[tx][r];
}

// ---------------------------------------------------------------- GEMM
// C[4096,2048] = A[4096,2048] * Bt^T (Bt pre-transposed [cols][k]).
// Double-buffered LDS tiles filled with global_load_async_to_lds_b128.
// mode 0: out_bf[B,N,Trows,D] (scaled) | mode 1: out_bf[B,N,D,S] | mode 2: f32
__global__ void __launch_bounds__(256) gemm_bf16_async(
    const __bf16* __restrict__ A, const __bf16* __restrict__ Bt,
    __bf16* __restrict__ out_bf, float* __restrict__ out_f32,
    float scale, int mode) {
  constexpr int BM = 128, BN = 128, BK = 32;
  __shared__ __bf16 As[2][BM][BK];   // 2 x 8 KB
  __shared__ __bf16 Bs[2][BN][BK];   // 2 x 8 KB

  const int tid  = threadIdx.x;
  const int wave = tid >> 5, lane = tid & 31;
  const int half = lane >> 4, l16 = lane & 15;
  const int wm = wave >> 2, wn = wave & 3;          // 2x4 wave grid
  const int m0 = blockIdx.y * BM, n0 = blockIdx.x * BN;

  auto issue_tiles = [&](int buf, int kt) {
#pragma unroll
    for (int i = 0; i < 2; ++i) {
      int c = i * 256 + tid;              // 512 16B chunks per tile
      int row = c >> 2, cir = c & 3;      // 4 chunks per 64B row
      async_b128(&As[buf][row][cir * 8],
                 A + (size_t)(m0 + row) * KdimE + kt + cir * 8);
    }
#pragma unroll
    for (int i = 0; i < 2; ++i) {
      int c = i * 256 + tid;
      int row = c >> 2, cir = c & 3;
      async_b128(&Bs[buf][row][cir * 8],
                 Bt + (size_t)(n0 + row) * KdimE + kt + cir * 8);
    }
  };

  v8f acc[4][2];
#pragma unroll
  for (int a = 0; a < 4; ++a)
#pragma unroll
    for (int b = 0; b < 2; ++b) acc[a][b] = zero8();

  constexpr int ITERS = KdimE / BK;   // 64
  issue_tiles(0, 0);
  int buf = 0;
  for (int it = 0; it < ITERS; ++it) {
    if (it + 1 < ITERS) {
      issue_tiles(buf ^ 1, (it + 1) * BK);   // prefetch next tile (4 ops)
      wait_asynccnt<4>();                    // oldest 4 (current tile) done
    } else {
      wait_asynccnt<0>();
    }
    __syncthreads();

    v16bf af[4];
#pragma unroll
    for (int mt = 0; mt < 4; ++mt) {
      const __bf16* p = &As[buf][wm * 64 + mt * 16 + l16][half * 8];
      af[mt] = load_frag(p, p + 16);          // A: K = h*8+0..7, h*8+16..23
    }
    v16bf bfr[2];
#pragma unroll
    for (int nt = 0; nt < 2; ++nt) {
      const __bf16* p = &Bs[buf][wn * 32 + nt * 16 + l16][half * 8];
      bfr[nt] = load_frag(p, p + 16);         // B: K = h*8+0..7, h*8+16..23
    }
#pragma unroll
    for (int mt = 0; mt < 4; ++mt)
#pragma unroll
      for (int nt = 0; nt < 2; ++nt)
        acc[mt][nt] = wmma_bf16(af[mt], bfr[nt], acc[mt][nt]);
    __syncthreads();
    buf ^= 1;
  }

#pragma unroll
  for (int mt = 0; mt < 4; ++mt)
#pragma unroll
    for (int nt = 0; nt < 2; ++nt)
#pragma unroll
      for (int i = 0; i < 8; ++i) {
        int gRow = m0 + wm * 64 + mt * 16 + half * 8 + i;
        int gCol = n0 + wn * 32 + nt * 16 + l16;
        float v = acc[mt][nt][i] * scale;
        if (mode == 2) {
          out_f32[(size_t)gRow * NcolsP + gCol] = v;
        } else {
          int bb = gRow >> 11, tt = gRow & 2047;
          int hn = gCol >> 7,  dd = gCol & 127;
          size_t dst = (mode == 0)
              ? (((size_t)(bb * NH + hn) * Tq) + tt) * DH + dd
              : (((size_t)(bb * NH + hn) * DH) + dd) * Ssz + tt;
          out_bf[dst] = f32_to_bf16(v);
        }
      }
}

// ---------------------------------------------------------------- flash attention
// Q: [B,N,T,D] bf16 (pre-scaled); K: [B,N,S,D]; V^T: [B,N,D,S]; O: [B,T,N,D].
// 4 waves x 16 query rows. K/V blocks double-buffered in LDS: the next
// s-block's async copies are issued before computing the current one, so
// global latency hides under the 32-WMMA compute phase.
__global__ void __launch_bounds__(128) flash_attn(
    const __bf16* __restrict__ Qp, const __bf16* __restrict__ Kp,
    const __bf16* __restrict__ Vt, __bf16* __restrict__ Op) {
  __shared__ __bf16 Kb[2][64][128];   // 2 x 16 KB (row = s, col = d)
  __shared__ __bf16 Vb[2][128][64];   // 2 x 16 KB (row = d, col = s)
  __shared__ __bf16 Pl[4][16][64];    //     8 KB  per-wave P tile

  const int tb = blockIdx.x, hn = blockIdx.y, bb = blockIdx.z;
  const int tid = threadIdx.x;
  const int wave = tid >> 5, lane = tid & 31;
  const int half = lane >> 4, l16 = lane & 15;

  const __bf16* Qh = Qp + (size_t)(bb * NH + hn) * Tq * DH;
  const __bf16* Kh = Kp + (size_t)(bb * NH + hn) * Ssz * DH;
  const __bf16* Vh = Vt + (size_t)(bb * NH + hn) * DH * Ssz;

  const int row0 = tb * 64 + wave * 16;

  auto stage_kv = [&](int buf, int sb) {
#pragma unroll
    for (int i = 0; i < 8; ++i) {           // K: 1024 16B chunks, 16/row
      int c = i * 128 + tid;
      int r = c >> 4, cir = c & 15;
      async_b128(&Kb[buf][r][cir * 8], Kh + (size_t)(sb + r) * DH + cir * 8);
    }
#pragma unroll
    for (int i = 0; i < 8; ++i) {           // V: 1024 16B chunks, 8/row
      int c = i * 128 + tid;
      int r = c >> 3, cir = c & 7;
      async_b128(&Vb[buf][r][cir * 8], Vh + (size_t)r * Ssz + sb + cir * 8);
    }
  };

  v16bf qf[4];
#pragma unroll
  for (int ks = 0; ks < 4; ++ks) {
    const __bf16* p = Qh + (size_t)(row0 + l16) * DH + ks * 32 + half * 8;
    qf[ks] = load_frag(p, p + 16);
  }

  v8f o[8];
#pragma unroll
  for (int dt = 0; dt < 8; ++dt) o[dt] = zero8();
  float mrow[8], lrow[8];
#pragma unroll
  for (int i = 0; i < 8; ++i) { mrow[i] = -1e30f; lrow[i] = 0.0f; }

  const int nblocks = tb + 1;
  stage_kv(0, 0);
  int buf = 0;
  for (int ib = 0; ib < nblocks; ++ib) {
    const int sb = ib * 64;
    if (ib + 1 < nblocks) {
      stage_kv(buf ^ 1, sb + 64);   // prefetch next s-block (16 ops)
      wait_asynccnt<16>();          // oldest 16 (current block) done
    } else {
      wait_asynccnt<0>();
    }
    __syncthreads();

    // ---- logits: S = Q * K^T  (16x64 per wave)
    v8f sacc[4];
#pragma unroll
    for (int nt = 0; nt < 4; ++nt) sacc[nt] = zero8();
#pragma unroll
    for (int ks = 0; ks < 4; ++ks)
#pragma unroll
      for (int nt = 0; nt < 4; ++nt) {
        const __bf16* kp = &Kb[buf][nt * 16 + l16][ks * 32 + half * 8];
        v16bf kf = load_frag(kp, kp + 16);
        sacc[nt] = wmma_bf16(qf[ks], kf, sacc[nt]);
      }

    // ---- causal mask (diagonal block only)
    if (ib == tb) {
#pragma unroll
      for (int nt = 0; nt < 4; ++nt)
#pragma unroll
        for (int i = 0; i < 8; ++i) {
          int s = sb + nt * 16 + l16;
          int t = row0 + half * 8 + i;
          if (s > t) sacc[nt][i] = -1e30f;
        }
    }

    // ---- online softmax (row = half*8 + i)
    float fi[8];
#pragma unroll
    for (int i = 0; i < 8; ++i) {
      float mx = fmaxf(fmaxf(sacc[0][i], sacc[1][i]), fmaxf(sacc[2][i], sacc[3][i]));
      mx = fmaxf(mx, __shfl_xor(mx, 1, 16));
      mx = fmaxf(mx, __shfl_xor(mx, 2, 16));
      mx = fmaxf(mx, __shfl_xor(mx, 4, 16));
      mx = fmaxf(mx, __shfl_xor(mx, 8, 16));
      float mnew = fmaxf(mrow[i], mx);
      fi[i] = __expf(mrow[i] - mnew);
      mrow[i] = mnew;
      float sum = 0.0f;
#pragma unroll
      for (int nt = 0; nt < 4; ++nt) {
        float p = __expf(sacc[nt][i] - mnew);
        Pl[wave][half * 8 + i][nt * 16 + l16] = f32_to_bf16(p);
        sum += p;
      }
      sum += __shfl_xor(sum, 1, 16);
      sum += __shfl_xor(sum, 2, 16);
      sum += __shfl_xor(sum, 4, 16);
      sum += __shfl_xor(sum, 8, 16);
      lrow[i] = lrow[i] * fi[i] + sum;
    }
#pragma unroll
    for (int dt = 0; dt < 8; ++dt)
#pragma unroll
      for (int i = 0; i < 8; ++i) o[dt][i] *= fi[i];

    // ---- O += P * V
#pragma unroll
    for (int ks2 = 0; ks2 < 2; ++ks2) {
      const __bf16* pp = &Pl[wave][l16][ks2 * 32 + half * 8];
      v16bf pa = load_frag(pp, pp + 16);
#pragma unroll
      for (int dt = 0; dt < 8; ++dt) {
        const __bf16* vp = &Vb[buf][dt * 16 + l16][ks2 * 32 + half * 8];
        v16bf vb = load_frag(vp, vp + 16);
        o[dt] = wmma_bf16(pa, vb, o[dt]);
      }
    }
    __syncthreads();   // all waves done with buf before it is restaged
    buf ^= 1;
  }

  // ---- normalize & store O[B,T,N,D]
#pragma unroll
  for (int dt = 0; dt < 8; ++dt)
#pragma unroll
    for (int i = 0; i < 8; ++i) {
      int t = row0 + half * 8 + i;
      int d = dt * 16 + l16;
      float v = o[dt][i] / lrow[i];
      Op[(((size_t)bb * Tq + t) * NH + hn) * DH + d] = f32_to_bf16(v);
    }
}

// ---------------------------------------------------------------- launch
extern "C" void kernel_launch(void* const* d_in, const int* in_sizes, int n_in,
                              void* d_out, int out_size, void* d_ws, size_t ws_size,
                              hipStream_t stream) {
  const float* Xq  = (const float*)d_in[0];
  const float* Xkv = (const float*)d_in[1];
  const float* Wq  = (const float*)d_in[2];
  const float* Wk  = (const float*)d_in[3];
  const float* Wv  = (const float*)d_in[4];
  const float* Wo  = (const float*)d_in[5];
  float* out = (float*)d_out;

  char* ws = (char*)d_ws;
  const size_t MB = 1024 * 1024;
  __bf16* Xq_bf  = (__bf16*)(ws + 0);        // 16 MB (reused for O_attn)
  __bf16* Xkv_bf = (__bf16*)(ws + 16 * MB);  // 16 MB
  __bf16* Wq_t   = (__bf16*)(ws + 32 * MB);  // 8 MB  [cols][k]
  __bf16* Wk_t   = (__bf16*)(ws + 40 * MB);  // 8 MB
  __bf16* Wv_t   = (__bf16*)(ws + 48 * MB);  // 8 MB
  __bf16* Wo_t   = (__bf16*)(ws + 56 * MB);  // 8 MB
  __bf16* Qp     = (__bf16*)(ws + 64 * MB);  // 16 MB [B,N,T,D]
  __bf16* Kp     = (__bf16*)(ws + 80 * MB);  // 16 MB [B,N,S,D]
  __bf16* Vtp    = (__bf16*)(ws + 96 * MB);  // 16 MB [B,N,D,S]
  __bf16* Oa     = Xq_bf;                    // reuse: [B,T,N,D]

  auto cvt = [&](const float* s, __bf16* d, int n) {
    int n4 = n / 4;
    cvt_f32_bf16<<<dim3((n4 + 255) / 256), dim3(256), 0, stream>>>(s, d, n4);
  };
  cvt(Xq,  Xq_bf,  Bsz * Tq * Emb);
  cvt(Xkv, Xkv_bf, Bsz * Ssz * Emb);

  dim3 tgrid(64, 64);
  cvt_transpose_bf16<<<tgrid, 256, 0, stream>>>(Wq, Wq_t);
  cvt_transpose_bf16<<<tgrid, 256, 0, stream>>>(Wk, Wk_t);
  cvt_transpose_bf16<<<tgrid, 256, 0, stream>>>(Wv, Wv_t);
  cvt_transpose_bf16<<<tgrid, 256, 0, stream>>>(Wo, Wo_t);

  dim3 ggrid(NcolsP / 128, Mrows / 128);
  float qscale = 1.0f / sqrtf((float)DH);
  gemm_bf16_async<<<ggrid, 256, 0, stream>>>(Xq_bf,  Wq_t, Qp,  nullptr, qscale, 0);
  gemm_bf16_async<<<ggrid, 256, 0, stream>>>(Xkv_bf, Wk_t, Kp,  nullptr, 1.0f,  0);
  gemm_bf16_async<<<ggrid, 256, 0, stream>>>(Xkv_bf, Wv_t, Vtp, nullptr, 1.0f,  1);

  flash_attn<<<dim3(Tq / 64, NH, Bsz), dim3(128), 0, stream>>>(Qp, Kp, Vtp, Oa);

  gemm_bf16_async<<<ggrid, 256, 0, stream>>>(Oa, Wo_t, nullptr, out, 1.0f, 2);
}
